// ArithmeticModel_86019605004851
// MI455X (gfx1250) — matrix operand order repaired and checked by reading, not compiled, not run on hardware
//
#include <hip/hip_runtime.h>

#define DMODEL 256
#define SEQ    1024
#define NB     8
#define NH     4
#define HDIM   64
#define FFDIM  1024
#define NLAYER 4
#define VOCAB  128
#define MROWS  (NB * SEQ)      // 8192 token rows

typedef unsigned short u16;
typedef __attribute__((ext_vector_type(16))) __bf16 v16bf;
typedef __attribute__((ext_vector_type(8)))  float  v8f;

union Frag {
    v16bf v;
    uint4 q[2];
};

__device__ inline u16 f2bf(float f) {
    unsigned int u = __float_as_uint(f);
    unsigned int r = u + 0x7fffu + ((u >> 16) & 1u);   // round-to-nearest-even
    return (u16)(r >> 16);
}

__device__ inline v8f vzero8() {
    v8f v;
#pragma unroll
    for (int i = 0; i < 8; ++i) v[i] = 0.0f;
    return v;
}

__device__ inline v8f wmma_bf16(const Frag& a, const Frag& b, v8f c) {
    return __builtin_amdgcn_wmma_f32_16x16x32_bf16(false, a.v, false, b.v,
                                                   (short)0, c, false, false);
}

// ---------------------------------------------------------------------------
// fp32 -> bf16 weight conversion
// ---------------------------------------------------------------------------
__global__ void f2bf_kernel(const float* __restrict__ in, u16* __restrict__ out, int n) {
    int i = blockIdx.x * blockDim.x + threadIdx.x;
    int stride = gridDim.x * blockDim.x;
    for (; i < n; i += stride) out[i] = f2bf(in[i]);
}

// ---------------------------------------------------------------------------
// Embedding: x = emb[id] * sqrt(D);  writes fp32 + bf16 shadow
// ---------------------------------------------------------------------------
__global__ __launch_bounds__(256)
void embed_kernel(const int* __restrict__ ids, const float* __restrict__ emb,
                  float* __restrict__ x, u16* __restrict__ xb) {
    int row = blockIdx.x;
    int d   = threadIdx.x;
    int id  = ids[row];
    float v = emb[(size_t)id * DMODEL + d] * 16.0f;   // sqrt(256)
    x [(size_t)row * DMODEL + d] = v;
    xb[(size_t)row * DMODEL + d] = f2bf(v);
}

// ---------------------------------------------------------------------------
// Generic bf16 WMMA GEMM: C[M,N] = A[M,K] @ B[K,N] + bias
// Block tile 128x128, 8 waves (4x2), each wave 32x64 = 2x4 WMMA tiles, K step 32.
// 8 WMMAs per K-chunk per wave; A/B staged through LDS (B stored K-contiguous).
// ---------------------------------------------------------------------------
template <bool RELU, bool OUTBF>
__global__ __launch_bounds__(256)
void gemm_bf16_kernel(const u16* __restrict__ A, const u16* __restrict__ Bw,
                      const float* __restrict__ bias, float* __restrict__ Cf,
                      u16* __restrict__ Cb, int M, int N, int K) {
    __shared__ __align__(16) u16 As[128 * 32];
    __shared__ __align__(16) u16 Bs[128 * 32];  // stored [n][k] (K contiguous)

    const int t      = threadIdx.x;
    const int lane   = t & 31;
    const int wave   = t >> 5;
    const int wm     = wave >> 1;       // 0..3 -> 32-row band
    const int wn     = wave & 1;        // 0..1 -> 64-col band
    const int l16    = lane & 15;
    const int half16 = lane >> 4;       // 0 or 1
    const int m0     = blockIdx.x * 128;
    const int n0     = blockIdx.y * 128;

    v8f acc[2][4];
#pragma unroll
    for (int i = 0; i < 2; ++i)
#pragma unroll
        for (int j = 0; j < 4; ++j) acc[i][j] = vzero8();

    for (int k0 = 0; k0 < K; k0 += 32) {
        __syncthreads();
        {   // stage A: 128 rows x 32 k, 2 threads per row, 32B each
            int row = t >> 1, h = t & 1;
            const uint4* src = (const uint4*)(A + (size_t)(m0 + row) * K + k0 + h * 16);
            uint4* dst = (uint4*)(As + row * 32 + h * 16);
            dst[0] = src[0];
            dst[1] = src[1];
        }
        {   // stage B transposed: Bs[n][k], 128 cols x 32 k
            int n   = t & 127;
            int k16 = (t >> 7) * 16;
#pragma unroll
            for (int i = 0; i < 16; ++i)
                Bs[n * 32 + k16 + i] = Bw[(size_t)(k0 + k16 + i) * N + n0 + n];
        }
        __syncthreads();

        Frag a[2];
#pragma unroll
        for (int i = 0; i < 2; ++i) {
            const u16* ap = As + (wm * 32 + 16 * i + l16) * 32;
            a[i].q[0] = *(const uint4*)(ap + half16 * 8);
            a[i].q[1] = *(const uint4*)(ap + 16 + half16 * 8);
        }
#pragma unroll
        for (int j = 0; j < 4; ++j) {
            Frag b;
            const u16* bp = Bs + (wn * 64 + 16 * j + l16) * 32 + half16 * 8;
            b.q[0] = *(const uint4*)(bp);
            b.q[1] = *(const uint4*)(bp + 16);
            acc[0][j] = wmma_bf16(a[0], b, acc[0][j]);
            acc[1][j] = wmma_bf16(a[1], b, acc[1][j]);
        }
    }

#pragma unroll
    for (int i = 0; i < 2; ++i) {
#pragma unroll
        for (int j = 0; j < 4; ++j) {
            int gm = m0 + wm * 32 + 16 * i + half16 * 8;
            int gn = n0 + wn * 64 + 16 * j + l16;
            float bv = bias ? bias[gn] : 0.0f;
#pragma unroll
            for (int r = 0; r < 8; ++r) {
                float v = acc[i][j][r] + bv;
                if (RELU) v = fmaxf(v, 0.0f);
                size_t off = (size_t)(gm + r) * N + gn;
                if (OUTBF) Cb[off] = f2bf(v);
                else       Cf[off] = v;
            }
        }
    }
}

// ---------------------------------------------------------------------------
// LayerNorm (+ optional residual add): one wave per 256-wide row.
// Writes fp32 (optional) and bf16 shadow.
// ---------------------------------------------------------------------------
__global__ __launch_bounds__(256)
void ln_kernel(const float* __restrict__ x, const float* __restrict__ y,
               const float* __restrict__ g, const float* __restrict__ bta,
               float* __restrict__ xout, u16* __restrict__ xb) {
    int lane = threadIdx.x & 31;
    int wave = threadIdx.x >> 5;
    size_t row = (size_t)blockIdx.x * 8 + wave;
    const float* xr = x + row * DMODEL;
    const float* yr = y ? (y + row * DMODEL) : nullptr;

    float v[8];
    float s = 0.0f, s2 = 0.0f;
#pragma unroll
    for (int i = 0; i < 8; ++i) {
        int d = lane + 32 * i;
        float tv = xr[d] + (yr ? yr[d] : 0.0f);
        v[i] = tv;
        s += tv;
        s2 += tv * tv;
    }
#pragma unroll
    for (int off = 16; off > 0; off >>= 1) {
        s  += __shfl_xor(s, off);
        s2 += __shfl_xor(s2, off);
    }
    float mean = s * (1.0f / DMODEL);
    float var  = s2 * (1.0f / DMODEL) - mean * mean;
    float inv  = rsqrtf(var + 1e-5f);
#pragma unroll
    for (int i = 0; i < 8; ++i) {
        int d = lane + 32 * i;
        float o = (v[i] - mean) * inv * g[d] + bta[d];
        if (xout) xout[row * DMODEL + d] = o;
        xb[row * DMODEL + d] = f2bf(o);
    }
}

// ---------------------------------------------------------------------------
// PoPE: mu = softplus(q/k), phase = s*freq + bias; builds bf16 qp/kp (feature
// dim 128 = [mu*cos | mu*sin]) and bf16 V.  Score scale folded into qp.
// ---------------------------------------------------------------------------
__global__ __launch_bounds__(64)
void pope_kernel(const float* __restrict__ q, const float* __restrict__ k,
                 const float* __restrict__ v, const float* __restrict__ pb,
                 u16* __restrict__ qp, u16* __restrict__ kp, u16* __restrict__ vb) {
    int d  = threadIdx.x;            // 0..63
    int s  = blockIdx.x;             // 0..S-1
    int bh = blockIdx.y;             // 0..B*H-1
    int b  = bh >> 2;
    int h  = bh & (NH - 1);

    size_t src = ((size_t)b * SEQ + s) * DMODEL + h * HDIM + d;
    float freq = __expf(-0.14391156855f * (float)d);       // 10000^(-d/64)
    float ph   = (float)s * freq + pb[h * HDIM + d];
    float sp, cp;
    __sincosf(ph, &sp, &cp);

    float qv = q[src], kv = k[src];
    float mq = (qv > 20.0f ? qv : log1pf(__expf(qv))) * 0.08838834764831845f; // 1/sqrt(128)
    float mk = (kv > 20.0f ? kv : log1pf(__expf(kv)));

    size_t dst = ((size_t)bh * SEQ + s) * (2 * HDIM) + d;
    qp[dst]        = f2bf(mq * cp);
    qp[dst + HDIM] = f2bf(mq * sp);
    kp[dst]        = f2bf(mk * cp);
    kp[dst + HDIM] = f2bf(mk * sp);
    vb[((size_t)bh * SEQ + s) * HDIM + d] = f2bf(v[src]);
}

// ---------------------------------------------------------------------------
// Flash attention with softmax1 (denominator 1 + sum(exp)).
// Grid: (S/128, B*H). Block: 256 threads = 8 waves, wave w owns 16 q-rows.
// Per 32-key iteration: 8 score WMMAs (feat K=128), online softmax, 4 PV WMMAs.
// ---------------------------------------------------------------------------
__global__ __launch_bounds__(256)
void attn_kernel(const u16* __restrict__ qp, const u16* __restrict__ kp,
                 const u16* __restrict__ vb, u16* __restrict__ out) {
    __shared__ __align__(16) u16 Ks [32 * 128];   // [key][feat]
    __shared__ __align__(16) u16 VsT[64 * 32];    // [dim][key]
    __shared__ __align__(16) u16 Ps [8 * 16 * 32];// per-wave P patch

    const int t      = threadIdx.x;
    const int lane   = t & 31;
    const int wave   = t >> 5;
    const int l16    = lane & 15;
    const int half16 = lane >> 4;
    const int bh     = blockIdx.y;
    const int b      = bh >> 2;
    const int h      = bh & (NH - 1);
    const int q0     = blockIdx.x * 128;
    const size_t sbase = (size_t)bh * SEQ;

    // Q fragments for this wave's 16 rows (A layout, 4 chunks of K=32 feats)
    const int qrow = q0 + wave * 16 + l16;
    Frag qa[4];
    const u16* qrp = qp + (sbase + qrow) * 128;
#pragma unroll
    for (int c = 0; c < 4; ++c) {
        qa[c].q[0] = *(const uint4*)(qrp + 32 * c + half16 * 8);
        qa[c].q[1] = *(const uint4*)(qrp + 32 * c + 16 + half16 * 8);
    }

    v8f acc[4];
#pragma unroll
    for (int nt = 0; nt < 4; ++nt) acc[nt] = vzero8();
    float mrun[8], lrun[8];
#pragma unroll
    for (int r = 0; r < 8; ++r) { mrun[r] = -1e30f; lrun[r] = 0.0f; }

    const int qibase = q0 + wave * 16 + half16 * 8;   // + r -> query index
    const int nkt = q0 / 32 + 4;                      // causal: keys <= q0+127

    for (int kt = 0; kt < nkt; ++kt) {
        const int ks0 = kt * 32;
        __syncthreads();
        {   // stage K tile: 32 keys x 128 feats
            int row = t >> 3, seg = t & 7;
            const uint4* src = (const uint4*)(kp + (sbase + ks0 + row) * 128 + seg * 16);
            uint4* dst = (uint4*)(Ks + row * 128 + seg * 16);
            dst[0] = src[0];
            dst[1] = src[1];
        }
        {   // stage V tile transposed: VsT[dim][key]
            int d = t & 63, kk = (t >> 6) * 8;
#pragma unroll
            for (int i = 0; i < 8; ++i)
                VsT[d * 32 + kk + i] = vb[(sbase + ks0 + kk + i) * HDIM + d];
        }
        __syncthreads();

        // scores: two 16-key tiles, K=128 features in 4 WMMA steps each
        v8f sc[2];
        sc[0] = vzero8();
        sc[1] = vzero8();
#pragma unroll
        for (int n = 0; n < 2; ++n) {
#pragma unroll
            for (int c = 0; c < 4; ++c) {
                Frag bk;
                const u16* kr = Ks + (n * 16 + l16) * 128 + 32 * c + half16 * 8;
                bk.q[0] = *(const uint4*)(kr);
                bk.q[1] = *(const uint4*)(kr + 16);
                sc[n] = wmma_bf16(qa[c], bk, sc[n]);
            }
        }

        // causal mask + online softmax1 update
        const int kj0 = ks0 + l16;
#pragma unroll
        for (int r = 0; r < 8; ++r) {
            int qi = qibase + r;
            float s0 = (kj0      > qi) ? -1e30f : sc[0][r];
            float s1 = (kj0 + 16 > qi) ? -1e30f : sc[1][r];
            float mx = fmaxf(s0, s1);
#pragma unroll
            for (int off = 1; off < 16; off <<= 1) mx = fmaxf(mx, __shfl_xor(mx, off));
            float mnew = fmaxf(mrun[r], mx);
            float fsc  = __expf(mrun[r] - mnew);
            float p0   = __expf(s0 - mnew);
            float p1   = __expf(s1 - mnew);
            float sum  = p0 + p1;
#pragma unroll
            for (int off = 1; off < 16; off <<= 1) sum += __shfl_xor(sum, off);
            lrun[r] = lrun[r] * fsc + sum;
            mrun[r] = mnew;
#pragma unroll
            for (int nt = 0; nt < 4; ++nt) acc[nt][r] *= fsc;
            int prow = r + 8 * half16;
            Ps[wave * 512 + prow * 32 + l16]      = f2bf(p0);
            Ps[wave * 512 + prow * 32 + 16 + l16] = f2bf(p1);
        }

        // P (16x32) as A-fragment, then PV: out(16x64) += P @ V
        Frag pa;
        const u16* pr = Ps + wave * 512 + l16 * 32;
        pa.q[0] = *(const uint4*)(pr + half16 * 8);
        pa.q[1] = *(const uint4*)(pr + 16 + half16 * 8);
#pragma unroll
        for (int nt = 0; nt < 4; ++nt) {
            Frag bv;
            const u16* vr = VsT + (nt * 16 + l16) * 32 + half16 * 8;
            bv.q[0] = *(const uint4*)(vr);
            bv.q[1] = *(const uint4*)(vr + 16);
            acc[nt] = wmma_bf16(pa, bv, acc[nt]);
        }
    }

    // epilogue: out = acc / (1 + l), write bf16 at (b, s, h*64 + d)
#pragma unroll
    for (int nt = 0; nt < 4; ++nt) {
        int d = nt * 16 + l16;
#pragma unroll
        for (int r = 0; r < 8; ++r) {
            int row = qibase + r;
            float o = acc[nt][r] / (1.0f + lrun[r]);
            out[((size_t)b * SEQ + row) * DMODEL + h * HDIM + d] = f2bf(o);
        }
    }
}

// ---------------------------------------------------------------------------
// Host-side orchestration
// ---------------------------------------------------------------------------
extern "C" void kernel_launch(void* const* d_in, const int* in_sizes, int n_in,
                              void* d_out, int out_size, void* d_ws, size_t ws_size,
                              hipStream_t stream) {
    (void)in_sizes; (void)n_in; (void)out_size; (void)ws_size;

    const int*   ids   = (const int*)  d_in[0];
    const float* emb   = (const float*)d_in[1];
    const float* Wq    = (const float*)d_in[2];
    const float* bq    = (const float*)d_in[3];
    const float* Wk    = (const float*)d_in[4];
    const float* bk    = (const float*)d_in[5];
    const float* Wv    = (const float*)d_in[6];
    const float* bv    = (const float*)d_in[7];
    const float* Wo    = (const float*)d_in[8];
    const float* bo    = (const float*)d_in[9];
    const float* pbias = (const float*)d_in[10];
    const float* W1    = (const float*)d_in[11];
    const float* b1    = (const float*)d_in[12];
    const float* W2    = (const float*)d_in[13];
    const float* b2    = (const float*)d_in[14];
    const float* ln1g  = (const float*)d_in[15];
    const float* ln1b  = (const float*)d_in[16];
    const float* ln2g  = (const float*)d_in[17];
    const float* ln2b  = (const float*)d_in[18];
    const float* lnfg  = (const float*)d_in[19];
    const float* lnfb  = (const float*)d_in[20];
    const float* Wlm   = (const float*)d_in[21];

    char* p = (char*)d_ws;
    auto alloc = [&](size_t bytes) -> char* {
        char* r = p;
        p += (bytes + 255) & ~(size_t)255;
        return r;
    };

    float* x    = (float*)alloc((size_t)MROWS * DMODEL * 4);
    u16*   xb   = (u16*)  alloc((size_t)MROWS * DMODEL * 2);
    float* qf   = (float*)alloc((size_t)MROWS * DMODEL * 4);
    float* kf   = (float*)alloc((size_t)MROWS * DMODEL * 4);
    float* vf   = (float*)alloc((size_t)MROWS * DMODEL * 4);
    float* yf   = (float*)alloc((size_t)MROWS * DMODEL * 4);
    u16*   qpb  = (u16*)  alloc((size_t)NB * NH * SEQ * 2 * HDIM * 2);
    u16*   kpb  = (u16*)  alloc((size_t)NB * NH * SEQ * 2 * HDIM * 2);
    u16*   vbb  = (u16*)  alloc((size_t)NB * NH * SEQ * HDIM * 2);
    u16*   atb  = (u16*)  alloc((size_t)MROWS * DMODEL * 2);
    u16*   hb   = (u16*)  alloc((size_t)MROWS * FFDIM * 2);
    u16*   wqb  = (u16*)  alloc((size_t)NLAYER * DMODEL * DMODEL * 2);
    u16*   wkb  = (u16*)  alloc((size_t)NLAYER * DMODEL * DMODEL * 2);
    u16*   wvb  = (u16*)  alloc((size_t)NLAYER * DMODEL * DMODEL * 2);
    u16*   wob  = (u16*)  alloc((size_t)NLAYER * DMODEL * DMODEL * 2);
    u16*   w1b  = (u16*)  alloc((size_t)NLAYER * DMODEL * FFDIM * 2);
    u16*   w2b  = (u16*)  alloc((size_t)NLAYER * FFDIM * DMODEL * 2);
    u16*   wlmb = (u16*)  alloc((size_t)DMODEL * VOCAB * 2);

    auto conv = [&](const float* src, u16* dst, int n) {
        int blocks = (n + 255) / 256;
        if (blocks > 2048) blocks = 2048;
        f2bf_kernel<<<blocks, 256, 0, stream>>>(src, dst, n);
    };
    conv(Wq,  wqb,  NLAYER * DMODEL * DMODEL);
    conv(Wk,  wkb,  NLAYER * DMODEL * DMODEL);
    conv(Wv,  wvb,  NLAYER * DMODEL * DMODEL);
    conv(Wo,  wob,  NLAYER * DMODEL * DMODEL);
    conv(W1,  w1b,  NLAYER * DMODEL * FFDIM);
    conv(W2,  w2b,  NLAYER * FFDIM * DMODEL);
    conv(Wlm, wlmb, DMODEL * VOCAB);

    embed_kernel<<<MROWS, 256, 0, stream>>>(ids, emb, x, xb);

    const dim3 gProj(MROWS / 128, DMODEL / 128);
    const dim3 gFF1 (MROWS / 128, FFDIM  / 128);
    const dim3 gLm  (MROWS / 128, VOCAB  / 128);
    const dim3 gAttn(SEQ / 128, NB * NH);
    const dim3 gPope(SEQ, NB * NH);

    for (int l = 0; l < NLAYER; ++l) {
        const u16* wq_l = wqb + (size_t)l * DMODEL * DMODEL;
        const u16* wk_l = wkb + (size_t)l * DMODEL * DMODEL;
        const u16* wv_l = wvb + (size_t)l * DMODEL * DMODEL;
        const u16* wo_l = wob + (size_t)l * DMODEL * DMODEL;
        const u16* w1_l = w1b + (size_t)l * DMODEL * FFDIM;
        const u16* w2_l = w2b + (size_t)l * FFDIM * DMODEL;

        gemm_bf16_kernel<false, false><<<gProj, 256, 0, stream>>>(
            xb, wq_l, bq + l * DMODEL, qf, nullptr, MROWS, DMODEL, DMODEL);
        gemm_bf16_kernel<false, false><<<gProj, 256, 0, stream>>>(
            xb, wk_l, bk + l * DMODEL, kf, nullptr, MROWS, DMODEL, DMODEL);
        gemm_bf16_kernel<false, false><<<gProj, 256, 0, stream>>>(
            xb, wv_l, bv + l * DMODEL, vf, nullptr, MROWS, DMODEL, DMODEL);

        pope_kernel<<<gPope, 64, 0, stream>>>(qf, kf, vf, pbias + l * NH * HDIM,
                                              qpb, kpb, vbb);

        attn_kernel<<<gAttn, 256, 0, stream>>>(qpb, kpb, vbb, atb);

        gemm_bf16_kernel<false, false><<<gProj, 256, 0, stream>>>(
            atb, wo_l, bo + l * DMODEL, yf, nullptr, MROWS, DMODEL, DMODEL);

        ln_kernel<<<MROWS / 8, 256, 0, stream>>>(x, yf, ln1g + l * DMODEL,
                                                 ln1b + l * DMODEL, x, xb);

        gemm_bf16_kernel<true, true><<<gFF1, 256, 0, stream>>>(
            xb, w1_l, b1 + l * FFDIM, nullptr, hb, MROWS, FFDIM, DMODEL);
        gemm_bf16_kernel<false, false><<<gProj, 256, 0, stream>>>(
            hb, w2_l, b2 + l * DMODEL, yf, nullptr, MROWS, DMODEL, FFDIM);

        ln_kernel<<<MROWS / 8, 256, 0, stream>>>(x, yf, ln2g + l * DMODEL,
                                                 ln2b + l * DMODEL, x, xb);
    }

    ln_kernel<<<MROWS / 8, 256, 0, stream>>>(x, nullptr, lnfg, lnfb, nullptr, xb);

    gemm_bf16_kernel<false, false><<<gLm, 256, 0, stream>>>(
        xb, wlmb, nullptr, (float*)d_out, nullptr, MROWS, VOCAB, DMODEL);
}
